// PartAttentionBlock_27169963114699
// MI455X (gfx1250) — matrix operand order repaired and checked by reading, not compiled, not run on hardware
//
#include <hip/hip_runtime.h>
#include <hip/hip_bf16.h>
#include <math.h>

// ---------------------------------------------------------------------------
// CDNA5 (gfx1250) wave32 WMMA implementation of the PartAttentionBlock.
// All GEMMs run on v_wmma_f32_16x16x32_bf16 (f32 accumulate).
// ---------------------------------------------------------------------------

typedef unsigned short u16;
typedef __attribute__((ext_vector_type(16))) __bf16 v16bf;
typedef __attribute__((ext_vector_type(16))) u16    v16u;
typedef __attribute__((ext_vector_type(8)))  u16    v8u;
typedef __attribute__((ext_vector_type(8)))  float  v8f;

#define FD    128      // temporal dim
#define KPN   27       // keypoints
#define HN    4        // heads
#define DIMD  256
#define WW    108      // tokens per window
#define MTOK  (1024 * 108)   // 110592 total tokens

// ---- helpers ---------------------------------------------------------------

__device__ __forceinline__ u16 f2b(float f) {           // f32 -> bf16 (RNE)
  unsigned int u = __builtin_bit_cast(unsigned int, f);
  unsigned int r = u + 0x7FFFu + ((u >> 16) & 1u);
  return (u16)(r >> 16);
}

__device__ __forceinline__ v8f vzero8() {
  v8f z;
#pragma unroll
  for (int e = 0; e < 8; ++e) z[e] = 0.f;
  return z;
}

// A fragment: 16x32 bf16, row m contiguous in K.
// lanes 0-15: K = {kb..kb+7, kb+16..kb+23}; lanes 16-31: +8.
__device__ __forceinline__ v16bf load_a_frag(const u16* base, int stride, int lane) {
  int m   = lane & 15;
  int klo = (lane & 16) ? 8 : 0;
  const u16* p = base + (size_t)m * stride + klo;
  v8u lo = *(const v8u*)(p);
  v8u hi = *(const v8u*)(p + 16);
  v16u u = __builtin_shufflevector(lo, hi, 0,1,2,3,4,5,6,7,8,9,10,11,12,13,14,15);
  return __builtin_bit_cast(v16bf, u);
}

// B fragment: 32x16 bf16, column n = lane&15 contiguous in K.
// lanes 0-15: K = {kb..kb+7, kb+16..kb+23}; lanes 16-31: +8.
__device__ __forceinline__ v16bf load_b_frag(const u16* base, int stride, int lane) {
  int n  = lane & 15;
  int k2 = (lane & 16) ? 8 : 0;
  const u16* p = base + (size_t)n * stride + k2;
  v8u lo = *(const v8u*)(p);
  v8u hi = *(const v8u*)(p + 16);
  v16u u = __builtin_shufflevector(lo, hi, 0,1,2,3,4,5,6,7,8,9,10,11,12,13,14,15);
  return __builtin_bit_cast(v16bf, u);
}

__device__ __forceinline__ v8f wmma_bf16(v16bf a, v16bf b, v8f c) {
  return __builtin_amdgcn_wmma_f32_16x16x32_bf16(false, a, false, b, (short)0, c,
                                                 false, false);
}

// ---- weight conversion -----------------------------------------------------

__global__ void cvt_f32_bf16(const float* __restrict__ src, u16* __restrict__ dst, int n) {
  int i = blockIdx.x * blockDim.x + threadIdx.x;
  if (i < n) dst[i] = f2b(src[i]);
}

// ---- LN1 + temporal shift + window partition -> bf16 activations -----------

__global__ void ln1_shift_kernel(const float* __restrict__ x,
                                 const float* __restrict__ g,
                                 const float* __restrict__ be,
                                 u16* __restrict__ A1) {
  int tid  = threadIdx.x;
  int r    = tid >> 2, part = tid & 3;
  int m    = blockIdx.x * 64 + r;
  int n    = m / WW, wq = m % WW;
  int b    = n >> 5, fw = n & 31;
  int tp   = wq / KPN, kp = wq % KPN;
  int t    = (fw * 4 + tp + 2) & (FD - 1);          // roll(x, -SHIFT, axis=time)
  const float* row = x + (((size_t)b * FD + t) * KPN + kp) * DIMD;

  int c0 = part * 64;
  float s = 0.f, q = 0.f;
#pragma unroll 4
  for (int c = 0; c < 64; ++c) { float v = row[c0 + c]; s += v; q += v * v; }
  s += __shfl_xor(s, 1, 32); s += __shfl_xor(s, 2, 32);
  q += __shfl_xor(q, 1, 32); q += __shfl_xor(q, 2, 32);
  float mean = s * (1.f / 256.f);
  float var  = q * (1.f / 256.f) - mean * mean;
  float rstd = rsqrtf(var + 1e-5f);

  u16* out = A1 + (size_t)m * DIMD;
  for (int c = 0; c < 64; ++c) {
    int cc = c0 + c;
    out[cc] = f2b((row[cc] - mean) * rstd * g[cc] + be[cc]);
  }
}

// ---- QKV GEMM: [110592,256] x [768,256]^T ; scatter into q/k/v^T bf16 ------

__global__ void gemm_qkv_kernel(const u16* __restrict__ A1, const u16* __restrict__ Wb,
                                const float* __restrict__ bias,
                                u16* __restrict__ Q, u16* __restrict__ Kb,
                                u16* __restrict__ VT) {
  int lane = threadIdx.x & 31, wid = threadIdx.x >> 5;
  int mbase = blockIdx.x * 64 + wid * 16;
  int nbase = blockIdx.y * 64;

  v8f acc[4];
#pragma unroll
  for (int j = 0; j < 4; ++j) acc[j] = vzero8();

  const u16* Abase = A1 + (size_t)mbase * DIMD;
#pragma unroll
  for (int kb = 0; kb < DIMD; kb += 32) {
    v16bf a = load_a_frag(Abase + kb, DIMD, lane);
#pragma unroll
    for (int j = 0; j < 4; ++j) {
      v16bf bf = load_b_frag(Wb + (size_t)(nbase + j * 16) * DIMD + kb, DIMD, lane);
      acc[j] = wmma_bf16(a, bf, acc[j]);
    }
  }

  int mrow0 = mbase + ((lane >> 4) << 3);
#pragma unroll
  for (int j = 0; j < 4; ++j) {
    int col = nbase + j * 16 + (lane & 15);    // 0..767
    int sec = col >> 8;                         // 0=q 1=k 2=v
    int hh  = (col >> 6) & 3;
    int d   = col & 63;
    float bv = bias[col];
#pragma unroll
    for (int r = 0; r < 8; ++r) {
      int m = mrow0 + r;
      int n = m / WW, wq = m % WW;
      size_t hb = (size_t)n * HN + hh;
      float val = acc[j][r] + bv;
      if (sec == 0)      Q [(hb * WW + wq) * 64 + d] = f2b(val * 0.125f);  // scale q
      else if (sec == 1) Kb[(hb * WW + wq) * 64 + d] = f2b(val);
      else               VT[(hb * 64 + d) * WW + wq] = f2b(val);           // v^T
    }
  }
}

// ---- attention: one block per (window, head); everything in LDS ------------

__global__ void attn_kernel(const u16* __restrict__ Q, const u16* __restrict__ Kmat,
                            const u16* __restrict__ VT, const float* __restrict__ eb,
                            u16* __restrict__ O) {
  __shared__ __align__(32) unsigned char smem[95232];
  u16*   qs  = (u16*)smem;               // [112][64]  14336 B
  u16*   ks  = (u16*)(smem + 14336);     // [112][64]  14336 B
  u16*   vts = (u16*)(smem + 28672);     // [64][128]  16384 B
  float* lg  = (float*)(smem + 45056);   // [112][112] 50176 B
  u16*   P   = (u16*)smem;               // [112][128] aliases qs/ks after S phase

  int tid = threadIdx.x, lane = tid & 31, wid = tid >> 5;
  int n = blockIdx.x >> 2, h = blockIdx.x & 3;
  size_t base = ((size_t)n * HN + h) * (size_t)(WW * 64);
  const u16* qg = Q + base;
  const u16* kg = Kmat + base;
  const u16* vg = VT + base;

  for (int i = tid; i < 112 * 64; i += 256) {
    int r = i >> 6, c = i & 63;
    u16 qv = 0, kv = 0;
    if (r < WW) { qv = qg[r * 64 + c]; kv = kg[r * 64 + c]; }
    qs[i] = qv; ks[i] = kv;
  }
  for (int i = tid; i < 64 * 128; i += 256) {
    int r = i >> 7, c = i & 127;
    vts[i] = (c < WW) ? vg[r * WW + c] : (u16)0;
  }
  __syncthreads();

  int fidx = n & 31;
  if (wid < 7) {                                   // S = q @ k^T  (112x112)
    v8f acc[7];
#pragma unroll
    for (int nt = 0; nt < 7; ++nt) acc[nt] = vzero8();
#pragma unroll
    for (int kb = 0; kb < 64; kb += 32) {
      v16bf a = load_a_frag(qs + wid * 16 * 64 + kb, 64, lane);
#pragma unroll
      for (int nt = 0; nt < 7; ++nt) {
        v16bf bf = load_b_frag(ks + nt * 16 * 64 + kb, 64, lane);
        acc[nt] = wmma_bf16(a, bf, acc[nt]);
      }
    }
#pragma unroll
    for (int nt = 0; nt < 7; ++nt) {
      int kw = nt * 16 + (lane & 15);
#pragma unroll
      for (int r = 0; r < 8; ++r) {
        int qw = wid * 16 + r + ((lane >> 4) << 3);
        float outv;
        if (qw < WW && kw < WW) {
          int tq = fidx * 4 + qw / KPN, tk = fidx * 4 + kw / KPN;
          int rq = (tq < 124) ? 0 : ((tq < 126) ? 1 : 2);
          int rk = (tk < 124) ? 0 : ((tk < 126) ? 1 : 2);
          float v = (rq == rk) ? acc[nt][r] * eb[qw * WW + kw] : 0.f;
          // masked_fill(attn==0, -1e4) then softmax(-attn): logit = -attn
          outv = (v == 0.f) ? 10000.f : -v;
        } else outv = -1e30f;
        lg[qw * 112 + kw] = outv;
      }
    }
  }
  __syncthreads();

  if (tid < 112) {                                 // row softmax -> P (bf16)
    float mx = -1e30f;
    for (int kw = 0; kw < WW; ++kw) mx = fmaxf(mx, lg[tid * 112 + kw]);
    float sum = 0.f;
    for (int kw = 0; kw < WW; ++kw) sum += __expf(lg[tid * 112 + kw] - mx);
    float inv = 1.f / sum;
    for (int kw = 0; kw < 128; ++kw)
      P[tid * 128 + kw] = (kw < WW) ? f2b(__expf(lg[tid * 112 + kw] - mx) * inv)
                                    : (u16)0;
  }
  __syncthreads();

  if (wid < 7) {                                   // O = P @ V  (112x64)
    v8f oacc[4];
#pragma unroll
    for (int nt = 0; nt < 4; ++nt) oacc[nt] = vzero8();
#pragma unroll
    for (int kb = 0; kb < 128; kb += 32) {
      v16bf a = load_a_frag(P + wid * 16 * 128 + kb, 128, lane);
#pragma unroll
      for (int nt = 0; nt < 4; ++nt) {
        v16bf bf = load_b_frag(vts + nt * 16 * 128 + kb, 128, lane);
        oacc[nt] = wmma_bf16(a, bf, oacc[nt]);
      }
    }
#pragma unroll
    for (int nt = 0; nt < 4; ++nt) {
      int d = nt * 16 + (lane & 15);
#pragma unroll
      for (int r = 0; r < 8; ++r) {
        int qw = wid * 16 + r + ((lane >> 4) << 3);
        if (qw < WW)
          O[((size_t)n * WW + qw) * DIMD + h * 64 + d] = f2b(oacc[nt][r]);
      }
    }
  }
}

// ---- proj GEMM + window roll(+SHIFT) + window_reverse + residual -----------

__global__ void gemm_proj_kernel(const u16* __restrict__ O, const u16* __restrict__ Wb,
                                 const float* __restrict__ bias,
                                 const float* __restrict__ xin,
                                 float* __restrict__ x2) {
  int lane = threadIdx.x & 31, wid = threadIdx.x >> 5;
  int mbase = blockIdx.x * 64 + wid * 16;
  int nbase = blockIdx.y * 64;

  v8f acc[4];
#pragma unroll
  for (int j = 0; j < 4; ++j) acc[j] = vzero8();

  const u16* Abase = O + (size_t)mbase * DIMD;
#pragma unroll
  for (int kb = 0; kb < DIMD; kb += 32) {
    v16bf a = load_a_frag(Abase + kb, DIMD, lane);
#pragma unroll
    for (int j = 0; j < 4; ++j) {
      v16bf bf = load_b_frag(Wb + (size_t)(nbase + j * 16) * DIMD + kb, DIMD, lane);
      acc[j] = wmma_bf16(a, bf, acc[j]);
    }
  }

  int mrow0 = mbase + ((lane >> 4) << 3);
#pragma unroll
  for (int j = 0; j < 4; ++j) {
    int col = nbase + j * 16 + (lane & 15);
    float bv = bias[col];
#pragma unroll
    for (int r = 0; r < 8; ++r) {
      int m = mrow0 + r;
      int n = m / WW, wq = m % WW;
      int dw = wq + 2; if (dw >= WW) dw -= WW;     // roll(o, +SHIFT, axis=W)
      int b = n >> 5, fw = n & 31;
      int t = fw * 4 + dw / KPN, kp = dw % KPN;    // window_reverse
      size_t idx = (((size_t)b * FD + t) * KPN + kp) * DIMD + col;
      x2[idx] = xin[idx] + acc[j][r] + bv;
    }
  }
}

// ---- fused LN2 + FC1 + GELU + FC2 + residual (hidden stays in LDS) ---------

__global__ void mlp_kernel(const float* __restrict__ x2, const float* __restrict__ g,
                           const float* __restrict__ be,
                           const u16* __restrict__ W1, const float* __restrict__ b1,
                           const u16* __restrict__ W2, const float* __restrict__ b2,
                           float* __restrict__ out) {
  __shared__ __align__(32) u16 Ash[64 * 256];   // 32 KB normalized activations
  __shared__ __align__(32) u16 Hsh[64 * 512];   // 64 KB hidden half-tile

  int tid = threadIdx.x, lane = tid & 31, wid = tid >> 5;
  int m0 = blockIdx.x * 64;

  { // LN2 into Ash
    int r = tid >> 2, part = tid & 3;
    const float* row = x2 + (size_t)(m0 + r) * DIMD;
    int c0 = part * 64;
    float s = 0.f, q = 0.f;
#pragma unroll 4
    for (int c = 0; c < 64; ++c) { float v = row[c0 + c]; s += v; q += v * v; }
    s += __shfl_xor(s, 1, 32); s += __shfl_xor(s, 2, 32);
    q += __shfl_xor(q, 1, 32); q += __shfl_xor(q, 2, 32);
    float mean = s * (1.f / 256.f);
    float var  = q * (1.f / 256.f) - mean * mean;
    float rstd = rsqrtf(var + 1e-5f);
    for (int c = 0; c < 64; ++c) {
      int cc = c0 + c;
      Ash[r * 256 + cc] = f2b((row[cc] - mean) * rstd * g[cc] + be[cc]);
    }
  }
  __syncthreads();

  int mi = wid & 3;            // 16-row stripe
  int nh = wid >> 2;           // 0/1: splits columns between wave pairs
  v8f oacc[2][4];              // fc2 accumulators persist across halves
#pragma unroll
  for (int ci = 0; ci < 2; ++ci)
#pragma unroll
    for (int j = 0; j < 4; ++j) oacc[ci][j] = vzero8();

  for (int half = 0; half < 2; ++half) {
    // FC1 + GELU for hidden cols [half*512, half*512+512) -> Hsh
    for (int ch = 0; ch < 4; ++ch) {
      int nbH = nh * 256 + ch * 64;            // column within half
      int nbG = half * 512 + nbH;              // global hidden column
      v8f acc[4];
#pragma unroll
      for (int j = 0; j < 4; ++j) acc[j] = vzero8();
#pragma unroll
      for (int kb = 0; kb < 256; kb += 32) {
        v16bf a = load_a_frag(Ash + mi * 16 * 256 + kb, 256, lane);
#pragma unroll
        for (int j = 0; j < 4; ++j) {
          v16bf bf = load_b_frag(W1 + (size_t)(nbG + j * 16) * 256 + kb, 256, lane);
          acc[j] = wmma_bf16(a, bf, acc[j]);
        }
      }
#pragma unroll
      for (int j = 0; j < 4; ++j) {
        int col = nbG + j * 16 + (lane & 15);
        int cl  = nbH + j * 16 + (lane & 15);
#pragma unroll
        for (int r = 0; r < 8; ++r) {
          int rr = mi * 16 + r + ((lane >> 4) << 3);
          float v = acc[j][r] + b1[col];
          v = 0.5f * v * (1.f + erff(v * 0.70710678118654752f));   // exact GELU
          Hsh[rr * 512 + cl] = f2b(v);
        }
      }
    }
    __syncthreads();

    // FC2 partial product over this K half
#pragma unroll 2
    for (int kb = 0; kb < 512; kb += 32) {
      v16bf a = load_a_frag(Hsh + mi * 16 * 512 + kb, 512, lane);
#pragma unroll
      for (int ci = 0; ci < 2; ++ci) {
        int nb2 = (nh * 2 + ci) * 64;
#pragma unroll
        for (int j = 0; j < 4; ++j) {
          v16bf bf = load_b_frag(W2 + (size_t)(nb2 + j * 16) * 1024 + half * 512 + kb,
                                 1024, lane);
          oacc[ci][j] = wmma_bf16(a, bf, oacc[ci][j]);
        }
      }
    }
    __syncthreads();
  }

  // epilogue: bias + residual -> out
#pragma unroll
  for (int ci = 0; ci < 2; ++ci) {
    int nb2 = (nh * 2 + ci) * 64;
#pragma unroll
    for (int j = 0; j < 4; ++j) {
      int col = nb2 + j * 16 + (lane & 15);
      float bv = b2[col];
#pragma unroll
      for (int r = 0; r < 8; ++r) {
        int m = m0 + mi * 16 + r + ((lane >> 4) << 3);
        size_t idx = (size_t)m * DIMD + col;
        out[idx] = oacc[ci][j][r] + bv + x2[idx];
      }
    }
  }
}

// ---------------------------------------------------------------------------

extern "C" void kernel_launch(void* const* d_in, const int* in_sizes, int n_in,
                              void* d_out, int out_size, void* d_ws, size_t ws_size,
                              hipStream_t stream) {
  const float* x      = (const float*)d_in[0];
  const float* eb     = (const float*)d_in[1];
  const float* w_qkv  = (const float*)d_in[2];
  const float* b_qkv  = (const float*)d_in[3];
  const float* w_proj = (const float*)d_in[4];
  const float* b_proj = (const float*)d_in[5];
  const float* g1     = (const float*)d_in[6];
  const float* be1    = (const float*)d_in[7];
  const float* g2     = (const float*)d_in[8];
  const float* be2    = (const float*)d_in[9];
  const float* w_fc1  = (const float*)d_in[10];
  const float* b_fc1  = (const float*)d_in[11];
  const float* w_fc2  = (const float*)d_in[12];
  const float* b_fc2  = (const float*)d_in[13];
  float* out = (float*)d_out;

  // workspace carve (~342 MB total)
  char* ws = (char*)d_ws;
  u16* wqkv_bf = (u16*)ws;  ws += (size_t)768 * 256 * 2;
  u16* wproj_bf = (u16*)ws; ws += (size_t)256 * 256 * 2;
  u16* wfc1_bf = (u16*)ws;  ws += (size_t)1024 * 256 * 2;
  u16* wfc2_bf = (u16*)ws;  ws += (size_t)256 * 1024 * 2;
  u16* A1  = (u16*)ws;      ws += (size_t)MTOK * 256 * 2;   // LN1 acts, reused as O
  u16* Qb  = (u16*)ws;      ws += (size_t)MTOK * 256 * 2;
  u16* Kb  = (u16*)ws;      ws += (size_t)MTOK * 256 * 2;
  u16* VTb = (u16*)ws;      ws += (size_t)MTOK * 256 * 2;
  float* x2 = (float*)ws;   ws += (size_t)MTOK * 256 * 4;

  cvt_f32_bf16<<<(196608 + 255) / 256, 256, 0, stream>>>(w_qkv,  wqkv_bf,  196608);
  cvt_f32_bf16<<<( 65536 + 255) / 256, 256, 0, stream>>>(w_proj, wproj_bf,  65536);
  cvt_f32_bf16<<<(262144 + 255) / 256, 256, 0, stream>>>(w_fc1,  wfc1_bf,  262144);
  cvt_f32_bf16<<<(262144 + 255) / 256, 256, 0, stream>>>(w_fc2,  wfc2_bf,  262144);

  ln1_shift_kernel<<<MTOK / 64, 256, 0, stream>>>(x, g1, be1, A1);
  gemm_qkv_kernel<<<dim3(MTOK / 64, 12), 128, 0, stream>>>(A1, wqkv_bf, b_qkv,
                                                           Qb, Kb, VTb);
  attn_kernel<<<4096, 256, 0, stream>>>(Qb, Kb, VTb, eb, A1 /* O reuses A1 */);
  gemm_proj_kernel<<<dim3(MTOK / 64, 4), 128, 0, stream>>>(A1, wproj_bf, b_proj,
                                                           x, x2);
  mlp_kernel<<<MTOK / 64, 256, 0, stream>>>(x2, g2, be2, wfc1_bf, b_fc1,
                                            wfc2_bf, b_fc2, out);
}
